// CausalSelfAttention_64690797412748
// MI455X (gfx1250) — matrix operand order repaired and checked by reading, not compiled, not run on hardware
//
#include <hip/hip_runtime.h>
#include <hip/hip_bf16.h>
#include <math.h>
#include <stdint.h>

typedef __attribute__((ext_vector_type(16))) __bf16 v16bf;
typedef __attribute__((ext_vector_type(8)))  __bf16 v8bf;
typedef __attribute__((ext_vector_type(8)))  float  v8f;

constexpr int Bc = 4;
constexpr int Tc = 2048;
constexpr int Cc = 1024;
constexpr int Hc = 16;
constexpr int Dc = 64;
constexpr int BT = Bc * Tc;          // 8192 rows
constexpr int MB = BT / 128;         // 64 m-blocks (128 rows each)
constexpr int NB = Cc / 128;         // 8 n-blocks (128 cols each)

__device__ __forceinline__ v8f wmma_bf16(v16bf a, v16bf b, v8f c) {
  return __builtin_amdgcn_wmma_f32_16x16x32_bf16(false, a, false, b, (short)0, c,
                                                 false, false);
}

// Per-lane async copy of 16 bytes global -> LDS (ASYNCcnt-tracked).
__device__ __forceinline__ void async_b128(uint32_t lds_byte_off, const void* gptr) {
  asm volatile("global_load_async_to_lds_b128 %0, %1, off"
               :: "v"(lds_byte_off), "v"(gptr) : "memory");
}

// A fragment 16x32 bf16 from row-major source. lane<16: K=0..7,16..23 ; lane>=16: K=8..15,24..31
__device__ __forceinline__ v16bf load_a_bf16(const __bf16* __restrict__ p0, int lda, int lane) {
  const int row = lane & 15;
  const int kb  = (lane & 16) ? 8 : 0;
  const __bf16* p = p0 + row * lda + kb;
  v16bf a;
#pragma unroll
  for (int i = 0; i < 8; ++i) {
    a[i]     = p[i];
    a[i + 8] = p[16 + i];
  }
  return a;
}

// bf16 B fragment, rows of source are K-major contiguous (B[k][n] = src[n][k])
__device__ __forceinline__ v16bf load_b_bf16_rm(const __bf16* __restrict__ p0, int ldn, int lane) {
  const int n  = lane & 15;
  const int kb = (lane & 16) ? 8 : 0;
  const __bf16* p = p0 + n * ldn + kb;
  v16bf b;
#pragma unroll
  for (int i = 0; i < 8; ++i) { b[i] = p[i]; b[i + 8] = p[16 + i]; }
  return b;
}

// ---------------- Kernel 0: fp32 -> bf16 conversion ----------------
__global__ void __launch_bounds__(256)
cvt_bf16_kernel(const float* __restrict__ src, __bf16* __restrict__ dst, int n) {
  const int i = (blockIdx.x * 256 + threadIdx.x) * 8;
  if (i < n) {
    const float4 f0 = *(const float4*)(src + i);
    const float4 f1 = *(const float4*)(src + i + 4);
    v8bf v;
    v[0] = (__bf16)f0.x; v[1] = (__bf16)f0.y; v[2] = (__bf16)f0.z; v[3] = (__bf16)f0.w;
    v[4] = (__bf16)f1.x; v[5] = (__bf16)f1.y; v[6] = (__bf16)f1.z; v[7] = (__bf16)f1.w;
    *(v8bf*)(dst + i) = v;
  }
}

// ---------------- Kernel 1: QKV projection + RoPE (128x128 block tiles, async LDS) ----------------
// Q,K stored [B,H,T,D]; V stored TRANSPOSED [B,H,D,T] so attention's PV B-fragments are contiguous.
__global__ void __launch_bounds__(256)
qkv_rope_kernel(const __bf16* __restrict__ xb,
                const __bf16* __restrict__ wqb, const __bf16* __restrict__ wkb,
                const __bf16* __restrict__ wvb,
                __bf16* __restrict__ Q, __bf16* __restrict__ K, __bf16* __restrict__ Vt) {
  __shared__ __bf16 At[2][128][32];
  __shared__ __bf16 Bt[2][128][32];

  const int tid  = threadIdx.x;
  const int lane = tid & 31;
  const int wv   = tid >> 5;
  const int rg   = wv & 3;              // row group: 32 rows
  const int cg   = wv >> 2;             // col group: 64 cols
  const int mat  = blockIdx.x / (MB * NB);
  const int rem  = blockIdx.x % (MB * NB);
  const int mb   = rem % MB;
  const int nb   = rem / MB;
  const int mrow0 = mb * 128;
  const int ncol0 = nb * 128;

  const __bf16* wb = (mat == 0) ? wqb : (mat == 1) ? wkb : wvb;

  auto issue = [&](int buf, int kc) {
#pragma unroll
    for (int i = 0; i < 2; ++i) {
      const int idx = tid + i * 256;          // 0..511
      const int row = idx >> 2;               // 0..127
      const int seg = (idx & 3) * 8;          // 0,8,16,24 (bf16 elems)
      async_b128((uint32_t)(uintptr_t)&At[buf][row][seg],
                 xb + (size_t)(mrow0 + row) * Cc + kc + seg);
      async_b128((uint32_t)(uintptr_t)&Bt[buf][row][seg],
                 wb + (size_t)(ncol0 + row) * Cc + kc + seg);
    }
  };

  v8f acc[2][4] = {};
  issue(0, 0);
  for (int kc = 0; kc < Cc; kc += 32) {
    const int cur = (kc >> 5) & 1;
    if (kc + 32 < Cc) {
      issue(cur ^ 1, kc + 32);
      asm volatile("s_wait_asynccnt 0x4" ::: "memory");   // prev tile done, next in flight
    } else {
      asm volatile("s_wait_asynccnt 0x0" ::: "memory");
    }
    __syncthreads();

    v16bf a[2];
#pragma unroll
    for (int q = 0; q < 2; ++q) {
      const int row = 32 * rg + 16 * q + (lane & 15);
      const int kb  = (lane & 16) ? 8 : 0;
#pragma unroll
      for (int i = 0; i < 8; ++i) {
        a[q][i]     = At[cur][row][kb + i];
        a[q][i + 8] = At[cur][row][kb + 16 + i];
      }
    }
#pragma unroll
    for (int t = 0; t < 4; ++t) {
      v16bf b;
      {
        const int n  = 64 * cg + t * 16 + (lane & 15);
        const int kb = (lane & 16) ? 8 : 0;
#pragma unroll
        for (int i = 0; i < 8; ++i) { b[i] = Bt[cur][n][kb + i]; b[i + 8] = Bt[cur][n][kb + 16 + i]; }
      }
#pragma unroll
      for (int q = 0; q < 2; ++q) acc[q][t] = wmma_bf16(a[q], b, acc[q][t]);
    }
    __syncthreads();
  }

  const int half   = lane >> 4;
  const int colloc = lane & 15;

#pragma unroll
  for (int q = 0; q < 2; ++q) {
    const int rbase = mrow0 + 32 * rg + 16 * q;
    if (mat < 2) {  // RoPE on Q and K
#pragma unroll
      for (int t = 0; t < 4; ++t) {
        const int col = ncol0 + 64 * cg + t * 16 + colloc;
        const int d   = col & 63;
        const float freq = __powf(10000.0f, -(float)(d & ~1) * (1.0f / 64.0f));
        const bool isEven = (d & 1) == 0;
#pragma unroll
        for (int r = 0; r < 8; ++r) {
          const int grow = rbase + r + 8 * half;
          const int tp   = grow & (Tc - 1);
          float sv, cv;
          __sincosf((float)tp * freq, &sv, &cv);
          float v  = acc[q][t][r];
          float pr = __shfl_xor(v, 1, 32);
          acc[q][t][r] = isEven ? (v * cv - pr * sv) : (pr * sv + v * cv);
        }
      }
    }
    if (mat == 0) {  // fold 1/sqrt(D) into Q
#pragma unroll
      for (int t = 0; t < 4; ++t) acc[q][t] = acc[q][t] * 0.125f;
    }

#pragma unroll
    for (int t = 0; t < 4; ++t) {
      const int col = ncol0 + 64 * cg + t * 16 + colloc;
      const int h   = col >> 6;
      const int d   = col & 63;
#pragma unroll
      for (int r = 0; r < 8; ++r) {
        const int grow = rbase + r + 8 * half;
        const int bb   = grow >> 11;        // / T
        const int tp   = grow & (Tc - 1);
        if (mat == 2) {
          // V transposed: [B,H,D,T]
          Vt[(((size_t)(bb * Hc + h) * Dc + d) * Tc) + tp] = (__bf16)acc[q][t][r];
        } else {
          __bf16* dst = (mat == 0) ? Q : K;
          dst[(((size_t)(bb * Hc + h) * Tc + tp) * Dc) + d] = (__bf16)acc[q][t][r];
        }
      }
    }
  }
}

// ---------------- Kernel 2: causal flash attention, bf16 WMMA ----------------
__global__ void __launch_bounds__(128)
flash_attn_kernel(const __bf16* __restrict__ Q, const __bf16* __restrict__ K,
                  const __bf16* __restrict__ Vt, __bf16* __restrict__ Y) {
  __shared__ __bf16 pbuf[4][16][32];

  const int lane = threadIdx.x & 31;
  const int wv   = threadIdx.x >> 5;
  const int w    = blockIdx.x * 4 + wv;
  const int qt   = w & 127;          // query tile within (b,h)
  const int bh   = w >> 7;
  const int b    = bh >> 4;
  const int h    = bh & 15;

  const size_t headoff = (size_t)(b * Hc + h) * Tc * Dc;
  const __bf16* Qb  = Q  + headoff;
  const __bf16* Kb  = K  + headoff;
  const __bf16* Vtb = Vt + headoff;   // [D][T] within head

  const int qr0   = qt * 16;
  const int qlast = qr0 + 15;
  const int half  = lane >> 4;
  const int col   = lane & 15;

  const v16bf qa0 = load_a_bf16(Qb + (size_t)qr0 * Dc + 0,  Dc, lane);
  const v16bf qa1 = load_a_bf16(Qb + (size_t)qr0 * Dc + 32, Dc, lane);

  float mrow[8], lrow[8];
#pragma unroll
  for (int r = 0; r < 8; ++r) { mrow[r] = -INFINITY; lrow[r] = 0.0f; }
  v8f oacc[4] = {};

  const int nchunks = (qlast + 32) / 32;  // 32-key chunks, causal
  for (int j = 0; j < nchunks; ++j) {
    const int k0 = j * 32;
    const bool have1 = (k0 + 16) <= qlast;

    v8f s0 = {}, s1 = {};
    {
      v16bf b0 = load_b_bf16_rm(Kb + (size_t)k0 * Dc + 0,  Dc, lane);
      s0 = wmma_bf16(qa0, b0, s0);
      v16bf b1 = load_b_bf16_rm(Kb + (size_t)k0 * Dc + 32, Dc, lane);
      s0 = wmma_bf16(qa1, b1, s0);
    }
    if (have1) {
      v16bf b0 = load_b_bf16_rm(Kb + (size_t)(k0 + 16) * Dc + 0,  Dc, lane);
      s1 = wmma_bf16(qa0, b0, s1);
      v16bf b1 = load_b_bf16_rm(Kb + (size_t)(k0 + 16) * Dc + 32, Dc, lane);
      s1 = wmma_bf16(qa1, b1, s1);
    }

    float alph[8];
#pragma unroll
    for (int r = 0; r < 8; ++r) {
      const int mloc = r + 8 * half;
      const int qrow = qr0 + mloc;
      float a0 = s0[r];
      if (k0 + col > qrow) a0 = -INFINITY;
      float a1 = have1 ? s1[r] : -INFINITY;
      if (k0 + 16 + col > qrow) a1 = -INFINITY;

      float rm = fmaxf(a0, a1);
      rm = fmaxf(rm, __shfl_xor(rm, 1, 32));
      rm = fmaxf(rm, __shfl_xor(rm, 2, 32));
      rm = fmaxf(rm, __shfl_xor(rm, 4, 32));
      rm = fmaxf(rm, __shfl_xor(rm, 8, 32));

      const float mnew  = fmaxf(mrow[r], rm);
      const float alpha = __expf(mrow[r] - mnew);
      const float e0 = __expf(a0 - mnew);
      const float e1 = __expf(a1 - mnew);

      float rs = e0 + e1;
      rs += __shfl_xor(rs, 1, 32);
      rs += __shfl_xor(rs, 2, 32);
      rs += __shfl_xor(rs, 4, 32);
      rs += __shfl_xor(rs, 8, 32);

      lrow[r] = lrow[r] * alpha + rs;
      mrow[r] = mnew;
      alph[r] = alpha;

      pbuf[wv][mloc][col]      = (__bf16)e0;
      pbuf[wv][mloc][16 + col] = (__bf16)e1;
    }

    // rescale running output
#pragma unroll
    for (int t = 0; t < 4; ++t)
#pragma unroll
      for (int r = 0; r < 8; ++r) oacc[t][r] *= alph[r];

    asm volatile("s_wait_dscnt 0x0" ::: "memory");

    // read P back as an A fragment (16x32) from LDS
    v16bf pa;
    {
      const int row = lane & 15;
      const int kb  = (lane & 16) ? 8 : 0;
#pragma unroll
      for (int i = 0; i < 8; ++i) {
        pa[i]     = pbuf[wv][row][kb + i];
        pa[i + 8] = pbuf[wv][row][kb + 16 + i];
      }
    }

    // PV: B[k][n] = V[k0+k][d=t*16+n] = Vt[d][k0+k]  -> contiguous reads
#pragma unroll
    for (int t = 0; t < 4; ++t) {
      v16bf bv = load_b_bf16_rm(Vtb + (size_t)(t * 16) * Tc + k0, Tc, lane);
      oacc[t] = wmma_bf16(pa, bv, oacc[t]);
    }
  }

  // normalize and store Y as bf16 [B*T, C]
#pragma unroll
  for (int t = 0; t < 4; ++t) {
    const int c = h * 64 + t * 16 + col;
#pragma unroll
    for (int r = 0; r < 8; ++r) {
      const int tq = qr0 + r + 8 * half;
      Y[((size_t)(b * Tc + tq)) * Cc + c] = (__bf16)(oacc[t][r] / lrow[r]);
    }
  }
}

// ---------------- Kernel 3: output projection Y(bf16) x Wo^T -> fp32 ----------------
__global__ void __launch_bounds__(256)
out_proj_kernel(const __bf16* __restrict__ Y, const __bf16* __restrict__ wob,
                float* __restrict__ out) {
  __shared__ __bf16 At[2][128][32];
  __shared__ __bf16 Bt[2][128][32];

  const int tid  = threadIdx.x;
  const int lane = tid & 31;
  const int wv   = tid >> 5;
  const int rg   = wv & 3;
  const int cg   = wv >> 2;
  const int mb   = blockIdx.x % MB;
  const int nb   = blockIdx.x / MB;
  const int mrow0 = mb * 128;
  const int ncol0 = nb * 128;

  auto issue = [&](int buf, int kc) {
#pragma unroll
    for (int i = 0; i < 2; ++i) {
      const int idx = tid + i * 256;
      const int row = idx >> 2;
      const int seg = (idx & 3) * 8;
      async_b128((uint32_t)(uintptr_t)&At[buf][row][seg],
                 Y + (size_t)(mrow0 + row) * Cc + kc + seg);
      async_b128((uint32_t)(uintptr_t)&Bt[buf][row][seg],
                 wob + (size_t)(ncol0 + row) * Cc + kc + seg);
    }
  };

  v8f acc[2][4] = {};
  issue(0, 0);
  for (int kc = 0; kc < Cc; kc += 32) {
    const int cur = (kc >> 5) & 1;
    if (kc + 32 < Cc) {
      issue(cur ^ 1, kc + 32);
      asm volatile("s_wait_asynccnt 0x4" ::: "memory");
    } else {
      asm volatile("s_wait_asynccnt 0x0" ::: "memory");
    }
    __syncthreads();

    v16bf a[2];
#pragma unroll
    for (int q = 0; q < 2; ++q) {
      const int row = 32 * rg + 16 * q + (lane & 15);
      const int kb  = (lane & 16) ? 8 : 0;
#pragma unroll
      for (int i = 0; i < 8; ++i) {
        a[q][i]     = At[cur][row][kb + i];
        a[q][i + 8] = At[cur][row][kb + 16 + i];
      }
    }
#pragma unroll
    for (int t = 0; t < 4; ++t) {
      v16bf b;
      {
        const int n  = 64 * cg + t * 16 + (lane & 15);
        const int kb = (lane & 16) ? 8 : 0;
#pragma unroll
        for (int i = 0; i < 8; ++i) { b[i] = Bt[cur][n][kb + i]; b[i + 8] = Bt[cur][n][kb + 16 + i]; }
      }
#pragma unroll
      for (int q = 0; q < 2; ++q) acc[q][t] = wmma_bf16(a[q], b, acc[q][t]);
    }
    __syncthreads();
  }

  const int half   = lane >> 4;
  const int colloc = lane & 15;
#pragma unroll
  for (int q = 0; q < 2; ++q) {
    const int rbase = mrow0 + 32 * rg + 16 * q;
#pragma unroll
    for (int t = 0; t < 4; ++t) {
      const int col = ncol0 + 64 * cg + t * 16 + colloc;
#pragma unroll
      for (int r = 0; r < 8; ++r) {
        const int grow = rbase + r + 8 * half;
        out[(size_t)grow * Cc + col] = acc[q][t][r];
      }
    }
  }
}

extern "C" void kernel_launch(void* const* d_in, const int* in_sizes, int n_in,
                              void* d_out, int out_size, void* d_ws, size_t ws_size,
                              hipStream_t stream) {
  const float* x  = (const float*)d_in[0];
  const float* wq = (const float*)d_in[1];
  const float* wk = (const float*)d_in[2];
  const float* wv = (const float*)d_in[3];
  const float* wo = (const float*)d_in[4];

  const size_t big = (size_t)BT * Cc;   // 8M elements
  const size_t wsz = (size_t)Cc * Cc;   // 1M elements
  __bf16* Q   = (__bf16*)d_ws;          // 16 MB each for Q,K,Vt,Y
  __bf16* K   = Q + big;
  __bf16* Vt  = K + big;
  __bf16* Y   = Vt + big;
  __bf16* xb  = Y + big;                // 16 MB
  __bf16* wqb = xb + big;               // 2 MB each
  __bf16* wkb = wqb + wsz;
  __bf16* wvb = wkb + wsz;
  __bf16* wob = wvb + wsz;
  float* out = (float*)d_out;

  // Kernel 0: fp32 -> bf16 conversions
  cvt_bf16_kernel<<<(int)(big / 2048), 256, 0, stream>>>(x, xb, (int)big);
  cvt_bf16_kernel<<<(int)(wsz / 2048), 256, 0, stream>>>(wq, wqb, (int)wsz);
  cvt_bf16_kernel<<<(int)(wsz / 2048), 256, 0, stream>>>(wk, wkb, (int)wsz);
  cvt_bf16_kernel<<<(int)(wsz / 2048), 256, 0, stream>>>(wv, wvb, (int)wsz);
  cvt_bf16_kernel<<<(int)(wsz / 2048), 256, 0, stream>>>(wo, wob, (int)wsz);

  // Kernel 1: 3 * 64 * 8 blocks of 256 threads (128x128 tiles)
  qkv_rope_kernel<<<3 * MB * NB, 256, 0, stream>>>(xb, wqb, wkb, wvb, Q, K, Vt);
  // Kernel 2: B*H*(T/16) = 8192 waves, 4 per block
  flash_attn_kernel<<<(Bc * Hc * (Tc / 16)) / 4, 128, 0, stream>>>(Q, K, Vt, Y);
  // Kernel 3: 64 * 8 blocks
  out_proj_kernel<<<MB * NB, 256, 0, stream>>>(Y, wob, out);
}
